// RotarySelfAttention_34239479284267
// MI455X (gfx1250) — matrix operand order repaired and checked by reading, not compiled, not run on hardware
//
#include <hip/hip_runtime.h>
#include <hip/hip_bf16.h>
#include <stdint.h>

// RotarySelfAttention for MI455X (gfx1250), wave32 + v_wmma_f32_16x16x32_f16
// + async global->LDS staging (ASYNCcnt) in the attention kernel.
// B=4, N=2048, DIM=512, H=8, DH=64, INNER=512.

typedef __attribute__((ext_vector_type(16))) _Float16 v16h;
typedef __attribute__((ext_vector_type(8)))  float    v8f;

#define B_      4
#define N_      2048
#define DIM_    512
#define H_      8
#define DH_     64
#define INNER_  512
#define QKVC_   1536   // 3 * INNER

__device__ __forceinline__ v8f wmma32(v16h a, v16h b, v8f c) {
  // D = A(16x32 f16) * B(32x16 f16) + C(16x16 f32)
  return __builtin_amdgcn_wmma_f32_16x16x32_f16(
      /*neg_a=*/false, a, /*neg_b=*/false, b,
      /*c_mod=*/(short)0, c, /*reuse_a=*/false, /*reuse_b=*/false);
}

// Async copy: 16 bytes per lane, global -> LDS, tracked by ASYNCcnt.
// One instruction per wave moves 512B with per-lane addresses.
__device__ __forceinline__ void async_copy_b128(uint32_t lds_byte_off,
                                                const void* gptr) {
  asm volatile("global_load_async_to_lds_b128 %0, %1, off"
               :: "v"(lds_byte_off), "v"((uint64_t)(uintptr_t)gptr)
               : "memory");
}

// ---- Fragment loaders (contiguous k order, same for A and B, wave32) ----------------
// A 16x32 (MxK) f16, row-major source with leading dim lda.
// lane<16: row=lane,  halves j -> K=j
// lane>=16: row=lane-16, halves j -> K=16+j
__device__ __forceinline__ v16h load_a_frag(const _Float16* A, int lda) {
  int lane = threadIdx.x & 31;
  int m  = lane & 15;
  int kb = (lane >> 4) << 4;                 // 0 or 16
  const _Float16* p = A + (size_t)m * lda + kb;
  v16h a;
#pragma unroll
  for (int j = 0; j < 8; ++j) a[j] = p[j];
#pragma unroll
  for (int j = 0; j < 8; ++j) a[8 + j] = p[8 + j];
  return a;
}

// B 32x16 (KxN) fragment from TRANSPOSED storage: Bt is NxK row-major (ld=ldt).
// Per-lane 16 contiguous halves -> b128 pairs (global or ds).
// lane<16: col=lane, halves j -> K=j;  lane>=16: col=lane-16, halves j -> K=16+j
__device__ __forceinline__ v16h load_b_frag_t(const _Float16* Bt, int ldt) {
  int lane = threadIdx.x & 31;
  int n  = lane & 15;
  int kb = (lane >> 4) << 4;                 // 0 or 16
  const _Float16* p = Bt + (size_t)n * ldt + kb;
  v16h b;
#pragma unroll
  for (int j = 0; j < 16; ++j) b[j] = p[j];
  return b;
}

// ---- Software-pipelined 16x64 wave-tile GEMM body, K = 512, lda = ldb = 512.
__device__ __forceinline__ void mm16x64_k512(const _Float16* __restrict__ Abase,
                                             const _Float16* __restrict__ Bbase,
                                             v8f c[4]) {
  v16h a  = load_a_frag(Abase, 512);
  v16h b0 = load_b_frag_t(Bbase + (size_t)(0 * 16) * 512, 512);
  v16h b1 = load_b_frag_t(Bbase + (size_t)(1 * 16) * 512, 512);
  v16h b2 = load_b_frag_t(Bbase + (size_t)(2 * 16) * 512, 512);
  v16h b3 = load_b_frag_t(Bbase + (size_t)(3 * 16) * 512, 512);
#pragma unroll
  for (int k = 32; k < 512; k += 32) {
    __builtin_prefetch(Abase + k + 32, 0, 1);
    __builtin_prefetch(Bbase + k + 32, 0, 1);
    v16h an  = load_a_frag(Abase + k, 512);
    v16h b0n = load_b_frag_t(Bbase + (size_t)(0 * 16) * 512 + k, 512);
    v16h b1n = load_b_frag_t(Bbase + (size_t)(1 * 16) * 512 + k, 512);
    v16h b2n = load_b_frag_t(Bbase + (size_t)(2 * 16) * 512 + k, 512);
    v16h b3n = load_b_frag_t(Bbase + (size_t)(3 * 16) * 512 + k, 512);
    c[0] = wmma32(a, b0, c[0]);
    c[1] = wmma32(a, b1, c[1]);
    c[2] = wmma32(a, b2, c[2]);
    c[3] = wmma32(a, b3, c[3]);
    a = an; b0 = b0n; b1 = b1n; b2 = b2n; b3 = b3n;
  }
  c[0] = wmma32(a, b0, c[0]);
  c[1] = wmma32(a, b1, c[1]);
  c[2] = wmma32(a, b2, c[2]);
  c[3] = wmma32(a, b3, c[3]);
}

// ---- Kernel 1: LayerNorm -> f16 --------------------------------------------
__global__ void ln_kernel(const float* __restrict__ x, const float* __restrict__ lnw,
                          const float* __restrict__ lnb, _Float16* __restrict__ xn) {
  int row = blockIdx.x;                       // 8192 rows
  int tid = threadIdx.x;                      // 256 threads
  const float* xr = x + (size_t)row * DIM_;
  float v0 = xr[tid], v1 = xr[tid + 256];
  float s  = v0 + v1;
  float s2 = v0 * v0 + v1 * v1;
#pragma unroll
  for (int off = 16; off > 0; off >>= 1) {
    s  += __shfl_xor(s,  off, 32);
    s2 += __shfl_xor(s2, off, 32);
  }
  __shared__ float ps[8], ps2[8];
  int warp = tid >> 5, lane = tid & 31;
  if (lane == 0) { ps[warp] = s; ps2[warp] = s2; }
  __syncthreads();
  float tot = 0.f, tot2 = 0.f;
#pragma unroll
  for (int i = 0; i < 8; ++i) { tot += ps[i]; tot2 += ps2[i]; }
  float mu  = tot * (1.0f / DIM_);
  float var = tot2 * (1.0f / DIM_) - mu * mu;
  float inv = rsqrtf(var + 1e-5f);
  _Float16* xo = xn + (size_t)row * DIM_;
  xo[tid]       = (_Float16)((v0 - mu) * inv * lnw[tid] + lnb[tid]);
  xo[tid + 256] = (_Float16)((v1 - mu) * inv * lnw[tid + 256] + lnb[tid + 256]);
}

// ---- Kernel 2: f32 -> f16 transpose-convert --------------------------------
__global__ void transpose_f16_kernel(const float* __restrict__ src,
                                     _Float16* __restrict__ dst, int K, int Ncol) {
  int i = blockIdx.x * blockDim.x + threadIdx.x;   // output-linear: coalesced writes
  if (i < K * Ncol) {
    int n = i / K;
    int k = i - n * K;
    dst[i] = (_Float16)src[(size_t)k * Ncol + n];
  }
}

// ---- Kernel 3: QKV GEMM (16x64 wave tile, WMMA) + RoPE epilogue ------------
__global__ void qkv_gemm_rope_kernel(const _Float16* __restrict__ xn,
                                     const _Float16* __restrict__ wT,
                                     const float* __restrict__ rope,
                                     _Float16* __restrict__ qh,
                                     _Float16* __restrict__ kh,
                                     _Float16* __restrict__ vt) {
  int warp = threadIdx.x >> 5;
  int tile = blockIdx.x * 4 + warp;           // 512 row-tiles * 24 col-groups = 12288
  int rowt = tile / 24;
  int colg = tile - rowt * 24;                // group of 64 output columns
  const _Float16* Abase = xn + (size_t)rowt * 16 * DIM_;
  const _Float16* Bbase = wT + (size_t)colg * 64 * DIM_;
  v8f c[4] = {{}, {}, {}, {}};
  mm16x64_k512(Abase, Bbase, c);
  int lane = threadIdx.x & 31;
  int hi   = lane >> 4;
  int nloc = lane & 15;
#pragma unroll
  for (int s = 0; s < 4; ++s) {
    int gcol = colg * 64 + s * 16 + nloc;
    int seg  = gcol >> 9;                     // 0=q, 1=k, 2=v
    int rem  = gcol & 511;
    int h    = rem >> 6;
    int d    = rem & 63;
    bool even = (d & 1) == 0;
#pragma unroll
    for (int r = 0; r < 8; ++r) {
      int row = rowt * 16 + r + 8 * hi;       // global row in [0, B*N)
      int bb  = row >> 11;
      int n   = row & (N_ - 1);
      float f  = rope[(size_t)row * DH_ + d];
      float cs = __cosf(f), sn = __sinf(f);
      float val = c[s][r];
      float pv  = __shfl_xor(val, 1, 32);     // pair partner (d^1 lives in lane^1)
      float ro  = even ? (val * cs - pv * sn) : (val * cs + pv * sn);
      _Float16 hv = (_Float16)ro;
      size_t bhn = (size_t)(bb * H_ + h) * N_ + n;
      if (seg == 0)      qh[bhn * DH_ + d] = hv;
      else if (seg == 1) kh[bhn * DH_ + d] = hv;
      else               vt[((size_t)(bb * H_ + h) * DH_ + d) * N_ + n] = hv;
    }
  }
}

// ---- Kernel 4: flash attention, async K/V staging to LDS -------------------
// All 4 waves of a block share one (b,h): K/V tiles are fetched ONCE per block
// via global_load_async_to_lds_b128 (double-buffered), then read as ds_load
// fragments. Each wave owns a private 16-row q-tile.
__global__ void attn_kernel(const _Float16* __restrict__ qh,
                            const _Float16* __restrict__ kh,
                            const _Float16* __restrict__ vt,
                            const float* __restrict__ rope,
                            _Float16* __restrict__ aout) {
  __shared__ _Float16 Kbuf[2][32 * 64];       // 32 seq rows x 64 d (4KB each)
  __shared__ _Float16 Vbuf[2][64 * 32];       // 64 d rows x 32 seq (4KB each)
  __shared__ _Float16 ldsP[4][16 * 32];       // per-wave P staging (C->A re-layout)
  int tid  = threadIdx.x;
  int warp = tid >> 5;
  int lane = tid & 31;
  int tile = blockIdx.x * 4 + warp;           // 32 bh * 128 row-tiles = 4096
  int bh = tile >> 7;                         // same bh for all 4 waves of a block
  int rt = tile & 127;
  int bb = bh >> 3, h = bh & 7;
  const _Float16* Q = qh + (size_t)bh * N_ * DH_;
  const _Float16* K = kh + (size_t)bh * N_ * DH_;
  const _Float16* V = vt + (size_t)bh * DH_ * N_;
  int row0 = rt * 16;
  int hi   = lane >> 4;
  int nloc = lane & 15;

  // Block-cooperative async stage of one 32-column K/V tile (4 asyncs per wave).
  auto issue_tile = [&](int bufi, int ct) {
#pragma unroll
    for (int q = 0; q < 2; ++q) {             // K tile: 256 chunks of 16B
      int c = q * 128 + tid;
      int row = c >> 3, part = c & 7;         // 8 chunks per 128B row
      uint32_t lds = (uint32_t)(uintptr_t)&Kbuf[bufi][row * 64 + part * 8];
      async_copy_b128(lds, K + (size_t)(ct + row) * DH_ + part * 8);
    }
#pragma unroll
    for (int q = 0; q < 2; ++q) {             // V tile: 256 chunks of 16B
      int c = q * 128 + tid;
      int d = c >> 2, part = c & 3;           // 4 chunks per 64B row
      uint32_t lds = (uint32_t)(uintptr_t)&Vbuf[bufi][d * 32 + part * 8];
      async_copy_b128(lds, V + (size_t)d * N_ + ct + part * 8);
    }
  };

  v16h aq0 = load_a_frag(Q + (size_t)row0 * DH_, DH_);       // K = 0..31
  v16h aq1 = load_a_frag(Q + (size_t)row0 * DH_ + 32, DH_);  // K = 32..63
  v8f o0 = {}, o1 = {}, o2 = {}, o3 = {};
  float m[8], l[8];
#pragma unroll
  for (int r = 0; r < 8; ++r) { m[r] = -1e30f; l[r] = 0.f; }
  _Float16* Pb = ldsP[warp];

  issue_tile(0, 0);                           // prologue: tile 0 in flight
  for (int ct = 0; ct < N_; ct += 32) {
    int cur = (ct >> 5) & 1;
    if (ct + 32 < N_) {
      issue_tile(cur ^ 1, ct + 32);           // stream next tile into other buffer
      asm volatile("s_wait_asynccnt 0x4" ::: "memory");  // current tile complete
    } else {
      asm volatile("s_wait_asynccnt 0x0" ::: "memory");
    }
    __syncthreads();                          // all waves' copies visible
    const _Float16* Kc = &Kbuf[cur][0];
    const _Float16* Vc = &Vbuf[cur][0];
    // K fragments from LDS (S = Q K^T; 16x32 tile = two subtiles, K-dim 64)
    v16h bk00 = load_b_frag_t(Kc, 64);
    v16h bk01 = load_b_frag_t(Kc + 32, 64);
    v16h bk10 = load_b_frag_t(Kc + 16 * 64, 64);
    v16h bk11 = load_b_frag_t(Kc + 16 * 64 + 32, 64);
    v16h bv0 = load_b_frag_t(Vc + (size_t)(0 * 16) * 32, 32);
    v16h bv1 = load_b_frag_t(Vc + (size_t)(1 * 16) * 32, 32);
    v16h bv2 = load_b_frag_t(Vc + (size_t)(2 * 16) * 32, 32);
    v16h bv3 = load_b_frag_t(Vc + (size_t)(3 * 16) * 32, 32);
    v8f c0 = {}, c1 = {};
    c0 = wmma32(aq0, bk00, c0);
    c0 = wmma32(aq1, bk01, c0);
    c1 = wmma32(aq0, bk10, c1);
    c1 = wmma32(aq1, bk11, c1);
    // online softmax: VGPR r holds row (r + 8*hi); reduce across 16-lane group
    float sc[8];
#pragma unroll
    for (int r = 0; r < 8; ++r) {
      float s0 = c0[r] * 0.125f;              // 1/sqrt(DH)
      float s1 = c1[r] * 0.125f;
      float mx = fmaxf(s0, s1);
      mx = fmaxf(mx, __shfl_xor(mx, 1, 32));
      mx = fmaxf(mx, __shfl_xor(mx, 2, 32));
      mx = fmaxf(mx, __shfl_xor(mx, 4, 32));
      mx = fmaxf(mx, __shfl_xor(mx, 8, 32));
      float mn = fmaxf(m[r], mx);
      float scale = __expf(m[r] - mn);
      float p0 = __expf(s0 - mn);
      float p1 = __expf(s1 - mn);
      float rs = p0 + p1;
      rs += __shfl_xor(rs, 1, 32);
      rs += __shfl_xor(rs, 2, 32);
      rs += __shfl_xor(rs, 4, 32);
      rs += __shfl_xor(rs, 8, 32);
      m[r] = mn;
      l[r] = l[r] * scale + rs;
      sc[r] = scale;
      Pb[(r + 8 * hi) * 32 + nloc]      = (_Float16)p0;
      Pb[(r + 8 * hi) * 32 + 16 + nloc] = (_Float16)p1;
    }
#pragma unroll
    for (int r = 0; r < 8; ++r) {
      o0[r] *= sc[r]; o1[r] *= sc[r]; o2[r] *= sc[r]; o3[r] *= sc[r];
    }
    v16h ap = load_a_frag(Pb, 32);            // P as A (16x32)
    o0 = wmma32(ap, bv0, o0);
    o1 = wmma32(ap, bv1, o1);
    o2 = wmma32(ap, bv2, o2);
    o3 = wmma32(ap, bv3, o3);
    __syncthreads();                          // reads done before buffer reuse
  }

  // epilogue: normalize by row-sum, inverse RoPE (cos same, sin negated), store
  v8f oo[4] = {o0, o1, o2, o3};
#pragma unroll
  for (int nt = 0; nt < 4; ++nt) {
    int d = nt * 16 + nloc;
    bool even = (d & 1) == 0;
#pragma unroll
    for (int r = 0; r < 8; ++r) {
      int n = row0 + r + 8 * hi;
      float val = oo[nt][r] / l[r];
      float pv  = __shfl_xor(val, 1, 32);
      float f   = rope[((size_t)bb * N_ + n) * DH_ + d];
      float cs = __cosf(f), sn = __sinf(f);
      float ro = even ? (val * cs + pv * sn) : (val * cs - pv * sn);
      aout[((size_t)bb * N_ + n) * INNER_ + h * DH_ + d] = (_Float16)ro;
    }
  }
}

// ---- Kernel 5: out GEMM (16x64 wave tile) + bias, f32 output ---------------
__global__ void out_gemm_kernel(const _Float16* __restrict__ a,
                                const _Float16* __restrict__ wT,
                                const float* __restrict__ bias,
                                float* __restrict__ out) {
  int warp = threadIdx.x >> 5;
  int tile = blockIdx.x * 4 + warp;           // 512 row-tiles * 8 col-groups = 4096
  int rowt = tile >> 3;
  int colg = tile & 7;
  const _Float16* Abase = a + (size_t)rowt * 16 * INNER_;
  const _Float16* Bbase = wT + (size_t)colg * 64 * INNER_;
  v8f c[4] = {{}, {}, {}, {}};
  mm16x64_k512(Abase, Bbase, c);
  int lane = threadIdx.x & 31;
  int hi = lane >> 4, nloc = lane & 15;
#pragma unroll
  for (int s = 0; s < 4; ++s) {
    int gcol = colg * 64 + s * 16 + nloc;
    float bv = bias[gcol];
#pragma unroll
    for (int r = 0; r < 8; ++r) {
      int row = rowt * 16 + r + 8 * hi;
      out[(size_t)row * DIM_ + gcol] = c[s][r] + bv;
    }
  }
}

// ---- Launcher ---------------------------------------------------------------
extern "C" void kernel_launch(void* const* d_in, const int* in_sizes, int n_in,
                              void* d_out, int out_size, void* d_ws, size_t ws_size,
                              hipStream_t stream) {
  (void)in_sizes; (void)n_in; (void)out_size; (void)ws_size;
  const float* x    = (const float*)d_in[0];   // (B,N,DIM)
  const float* rope = (const float*)d_in[1];   // (B,N,DH)
  const float* lnw  = (const float*)d_in[2];
  const float* lnb  = (const float*)d_in[3];
  const float* wqkv = (const float*)d_in[4];   // (DIM, 3*INNER)
  const float* wout = (const float*)d_in[5];   // (INNER, DIM)
  const float* bout = (const float*)d_in[6];
  float* out = (float*)d_out;

  char* ws = (char*)d_ws;
  size_t off = 0;
  auto carve = [&](size_t bytes) -> void* {
    void* p = ws + off;
    off += (bytes + 255) & ~(size_t)255;
    return p;
  };
  _Float16* xn     = (_Float16*)carve((size_t)B_ * N_ * DIM_ * 2);      // 8 MB
  _Float16* wqkvT  = (_Float16*)carve((size_t)QKVC_ * DIM_ * 2);        // 1.5 MB
  _Float16* woutT  = (_Float16*)carve((size_t)DIM_ * INNER_ * 2);       // 0.5 MB
  _Float16* qh     = (_Float16*)carve((size_t)B_ * H_ * N_ * DH_ * 2);  // 8 MB
  _Float16* kh     = (_Float16*)carve((size_t)B_ * H_ * N_ * DH_ * 2);  // 8 MB
  _Float16* vt     = (_Float16*)carve((size_t)B_ * H_ * DH_ * N_ * 2);  // 8 MB
  _Float16* aout   = (_Float16*)carve((size_t)B_ * N_ * INNER_ * 2);    // 8 MB

  ln_kernel<<<B_ * N_, 256, 0, stream>>>(x, lnw, lnb, xn);
  transpose_f16_kernel<<<(DIM_ * QKVC_ + 255) / 256, 256, 0, stream>>>(wqkv, wqkvT, DIM_, QKVC_);
  transpose_f16_kernel<<<(INNER_ * DIM_ + 255) / 256, 256, 0, stream>>>(wout, woutT, INNER_, DIM_);
  qkv_gemm_rope_kernel<<<3072, 128, 0, stream>>>(xn, wqkvT, rope, qh, kh, vt);
  attn_kernel<<<1024, 128, 0, stream>>>(qh, kh, vt, rope, aout);
  out_gemm_kernel<<<1024, 128, 0, stream>>>(aout, woutT, bout, out);
}
